// SelfAttention_57509612093767
// MI455X (gfx1250) — matrix operand order repaired and checked by reading, not compiled, not run on hardware
//
#include <hip/hip_runtime.h>
#include <hip/hip_bf16.h>

typedef __attribute__((ext_vector_type(16))) __bf16 v16bf;
typedef __attribute__((ext_vector_type(8)))  float  v8f;

#define DIMK   1024
#define NHEADS 16
#define HDIM   64
#define SEQ    2048
#define BATCH  2
#define MROWS  4096   // BATCH * SEQ

__device__ __forceinline__ unsigned short f2bf(float f) {
    unsigned int u = __float_as_uint(f);
    u += 0x7FFFu + ((u >> 16) & 1u);   // round-to-nearest-even
    return (unsigned short)(u >> 16);
}

union FragAB {
    v16bf v;
    uint4 q[2];        // two aligned 16-byte halves -> ds_load_b128 each
    unsigned int u[8];
};

__device__ __forceinline__ uint4 lds128(const unsigned short* p) {
    return *reinterpret_cast<const uint4*>(p);
}

// ---- DPP16 row-rotate reductions: full reduce over each 16-lane half -------
// (DPP rows == the 16-lane halves that hold accumulator rows r / r+8)
template<int CTRL>
__device__ __forceinline__ float dpp_rot(float x) {
    return __int_as_float(__builtin_amdgcn_update_dpp(
        0, __float_as_int(x), CTRL, 0xf, 0xf, true));
}
__device__ __forceinline__ float rowmax16(float v) {
    v = fmaxf(v, dpp_rot<0x128>(v));   // row_ror:8
    v = fmaxf(v, dpp_rot<0x124>(v));   // row_ror:4
    v = fmaxf(v, dpp_rot<0x122>(v));   // row_ror:2
    v = fmaxf(v, dpp_rot<0x121>(v));   // row_ror:1
    return v;
}
__device__ __forceinline__ float rowsum16(float v) {
    v += dpp_rot<0x128>(v);
    v += dpp_rot<0x124>(v);
    v += dpp_rot<0x122>(v);
    v += dpp_rot<0x121>(v);
    return v;
}

// ---- LDS XOR swizzles: all multiples of 8 elements (16 B) so every fragment
// half stays a contiguous, 16B-aligned block.
__device__ __forceinline__ int swzA(int row) { return (row & 3) << 3; }              // As: k^
__device__ __forceinline__ int swzB(int n)   { return (((n >> 3) ^ n) & 3) << 3; }   // BsT: k^
__device__ __forceinline__ int swzQ(int row) { return (row & 7) << 3; }              // Qs/Ks: d^
__device__ __forceinline__ int swzV(int d)   { return (d & 3) << 3; }                // VsT: key^
__device__ __forceinline__ int swzP(int m)   { return (m & 3) << 3; }                // Ps: k^

__global__ void cvt_f32_bf16(const float* __restrict__ in,
                             unsigned short* __restrict__ out, int n) {
    for (int i = blockIdx.x * blockDim.x + threadIdx.x; i < n;
         i += gridDim.x * blockDim.x)
        out[i] = f2bf(in[i]);
}

// ---------------------------------------------------------------------------
// Tiled bf16 WMMA GEMM, double-buffered LDS. Tile 128x128, BK=32, 8 waves.
// MODE 0: bf16 output scattered to (b, h, s, d) head layout (QKV projections)
// MODE 1: f32 output row-major (final output projection)
// ---------------------------------------------------------------------------
#define BM 128
#define BN 128
#define BK 32
#define NKT (DIMK / BK)

template<int MODE>
__global__ __launch_bounds__(256) void gemm_bf16_wmma(
    const unsigned short* __restrict__ A,
    const unsigned short* __restrict__ Bw,
    const float* __restrict__ bias,
    unsigned short* __restrict__ outB,
    float* __restrict__ outF)
{
    __shared__ __align__(16) unsigned short As[2][BM * BK];    // 2 x 8 KB
    __shared__ __align__(16) unsigned short BsT[2][BN * BK];   // 2 x 8 KB [n][k]

    const int tid  = threadIdx.x;
    const int lane = tid & 31;
    const int wave = tid >> 5;
    const int hi   = lane >> 4;
    const int l15  = lane & 15;
    const int wm   = (wave & 3) * 32;
    const int wn   = (wave >> 2) * 64;
    const int m0   = blockIdx.x * BM;
    const int n0   = blockIdx.y * BN;

    v8f acc[2][4];
#pragma unroll
    for (int t = 0; t < 2; ++t)
#pragma unroll
        for (int u = 0; u < 4; ++u)
#pragma unroll
            for (int r = 0; r < 8; ++r) acc[t][u][r] = 0.0f;

    uint4 ra[2], rb[2];
    auto fetch = [&](int kt) {
        const int k0 = kt * BK;
#pragma unroll
        for (int it = 0; it < 2; ++it) {
            const int i  = tid + it * 256;
            const int r1 = i >> 2, ca = (i & 3) << 3;
            ra[it] = *reinterpret_cast<const uint4*>(
                &A[(size_t)(m0 + r1) * DIMK + k0 + ca]);
            const int r2 = i >> 4, cb = (i & 15) << 3;
            rb[it] = *reinterpret_cast<const uint4*>(
                &Bw[(size_t)(k0 + r2) * DIMK + n0 + cb]);
        }
    };
    auto stage = [&](int buf) {
#pragma unroll
        for (int it = 0; it < 2; ++it) {
            const int i  = tid + it * 256;
            const int r1 = i >> 2, ca = (i & 3) << 3;
            *reinterpret_cast<uint4*>(&As[buf][r1 * BK + (ca ^ swzA(r1))]) = ra[it];
            const int r2 = i >> 4, cb = (i & 15) << 3;
            unsigned short tmp[8];
            __builtin_memcpy(tmp, &rb[it], 16);
#pragma unroll
            for (int j = 0; j < 8; ++j) {
                const int n = cb + j;
                BsT[buf][n * BK + (r2 ^ swzB(n))] = tmp[j];
            }
        }
    };

    fetch(0);
    stage(0);
    __syncthreads();

    for (int kt = 0; kt < NKT; ++kt) {
        const int cur = kt & 1;
        if (kt + 1 < NKT) fetch(kt + 1);   // overlap next tile with compute

        // A fragments: halves are 16B blocks at (8hi)^s and (16+8hi)^s
        FragAB af[2];
#pragma unroll
        for (int t = 0; t < 2; ++t) {
            const int row = wm + t * 16 + l15;
            const int sw  = swzA(row);
            const unsigned short* rp = &As[cur][row * BK];
            af[t].q[0] = lds128(rp + ((8 * hi) ^ sw));
            af[t].q[1] = lds128(rp + ((16 + 8 * hi) ^ sw));
        }
#pragma unroll
        for (int u = 0; u < 4; ++u) {
            // B fragment: halves at (8hi)^s and (16+8hi)^s
            FragAB bf;
            const int n  = wn + u * 16 + l15;
            const int sw = swzB(n);
            const unsigned short* np = &BsT[cur][n * BK];
            bf.q[0] = lds128(np + ((8 * hi) ^ sw));
            bf.q[1] = lds128(np + ((16 + 8 * hi) ^ sw));
#pragma unroll
            for (int t = 0; t < 2; ++t)
                acc[t][u] = __builtin_amdgcn_wmma_f32_16x16x32_bf16(
                    false, af[t].v, false, bf.v, (short)0, acc[t][u], false, false);
        }

        if (kt + 1 < NKT) stage((kt + 1) & 1);
        __syncthreads();
    }

    // epilogue: C/D layout -> element (row = r + 8*hi, col = l15) in VGPR r
#pragma unroll
    for (int u = 0; u < 4; ++u) {
        const int col = n0 + wn + u * 16 + l15;
        const float bv = bias[col];
#pragma unroll
        for (int t = 0; t < 2; ++t) {
#pragma unroll
            for (int r = 0; r < 8; ++r) {
                const int row = m0 + wm + t * 16 + r + 8 * hi;
                const float val = acc[t][u][r] + bv;
                if (MODE == 0) {
                    const int h = col >> 6, d = col & (HDIM - 1);
                    const int b = row >> 11, s = row & (SEQ - 1);
                    outB[(((size_t)b * NHEADS + h) * SEQ + s) * HDIM + d] = f2bf(val);
                } else {
                    outF[(size_t)row * DIMK + col] = val;
                }
            }
        }
    }
}

// ---------------------------------------------------------------------------
// Flash attention: grid (SEQ/128, BATCH*NHEADS), 256 threads (8 waves).
// Wave w owns a 16-query tile; 32-key blocks double-buffered in LDS.
// 8 v_wmma per key block; online softmax fully on VALU (DPP reductions).
// ---------------------------------------------------------------------------
#define NKB (SEQ / 32)

__global__ __launch_bounds__(256) void flash_attn_wmma(
    const unsigned short* __restrict__ Q,   // (B*H, S, 64) bf16
    const unsigned short* __restrict__ K,
    const unsigned short* __restrict__ V,
    const unsigned char* __restrict__ mask, // (B, S) bool
    unsigned short* __restrict__ ctx)       // (B*S, DIMK) bf16
{
    __shared__ __align__(16) unsigned short Qs[128 * 64];      // 16 KB [q][d]
    __shared__ __align__(16) unsigned short Ks[2][32 * 64];    // 2x4 KB [key][d]
    __shared__ __align__(16) unsigned short VsT[2][64 * 32];   // 2x4 KB [d][key]
    __shared__ __align__(16) unsigned short Ps[8][16 * 32];    // 8 KB wave-private

    const int tid  = threadIdx.x;
    const int lane = tid & 31;
    const int wave = tid >> 5;
    const int hi   = lane >> 4;
    const int l15  = lane & 15;
    const int bh   = blockIdx.y;
    const int b    = bh >> 4;
    const int h    = bh & 15;
    const int q0   = blockIdx.x * 128;

    const unsigned short* Qg = Q + (size_t)bh * SEQ * HDIM;
    const unsigned short* Kg = K + (size_t)bh * SEQ * HDIM;
    const unsigned short* Vg = V + (size_t)bh * SEQ * HDIM;

    const int rkv = tid >> 3;            // key row this thread stages
    const int ckv = (tid & 7) << 3;      // d chunk this thread stages

    for (int i = tid; i < (128 * 64) / 8; i += 256) {
        const int r = i >> 3, c8 = (i & 7) << 3;
        *reinterpret_cast<uint4*>(&Qs[r * 64 + (c8 ^ swzQ(r))]) =
            *reinterpret_cast<const uint4*>(&Qg[(size_t)(q0 + r) * 64 + c8]);
    }

    uint4 rk, rv;
    unsigned char mcur[2], mnxt[2];
    auto fetchKV = [&](int kb) {
        rk = *reinterpret_cast<const uint4*>(&Kg[(size_t)(kb * 32 + rkv) * 64 + ckv]);
        rv = *reinterpret_cast<const uint4*>(&Vg[(size_t)(kb * 32 + rkv) * 64 + ckv]);
    };
    auto stageKV = [&](int buf) {
        *reinterpret_cast<uint4*>(&Ks[buf][rkv * 64 + (ckv ^ swzQ(rkv))]) = rk;
        unsigned short tmp[8];
        __builtin_memcpy(tmp, &rv, 16);
#pragma unroll
        for (int j = 0; j < 8; ++j) {
            const int d = ckv + j;
            VsT[buf][d * 32 + (rkv ^ swzV(d))] = tmp[j];
        }
    };

    fetchKV(0);
    stageKV(0);
    mcur[0] = mask[(size_t)b * SEQ + l15];
    mcur[1] = mask[(size_t)b * SEQ + 16 + l15];
    __syncthreads();

    // Q A-fragments (rows wave*16 + l15, K-dim = d split into two 32-chunks)
    FragAB qa[2];
    {
        const int row = wave * 16 + l15;
        const int sw  = swzQ(row);
        const unsigned short* rp = &Qs[row * 64];
#pragma unroll
        for (int f = 0; f < 2; ++f) {
            qa[f].q[0] = lds128(rp + ((32 * f + 8 * hi) ^ sw));
            qa[f].q[1] = lds128(rp + ((32 * f + 16 + 8 * hi) ^ sw));
        }
    }

    float mrow[8], lrow[8];
    v8f o[4];
#pragma unroll
    for (int r = 0; r < 8; ++r) { mrow[r] = -3.0e38f; lrow[r] = 0.0f; }
#pragma unroll
    for (int u = 0; u < 4; ++u)
#pragma unroll
        for (int r = 0; r < 8; ++r) o[u][r] = 0.0f;

    const float scale = 0.125f;   // 1/sqrt(64)
    const float NEG   = -3.0e38f;

    for (int kb = 0; kb < NKB; ++kb) {
        const int cur = kb & 1;
        if (kb + 1 < NKB) {                       // overlap next block fetch
            fetchKV(kb + 1);
            mnxt[0] = mask[(size_t)b * SEQ + (kb + 1) * 32 + l15];
            mnxt[1] = mask[(size_t)b * SEQ + (kb + 1) * 32 + 16 + l15];
            if (kb + 2 < NKB) {
                __builtin_prefetch(&Kg[(size_t)((kb + 2) * 32 + rkv) * 64 + ckv], 0, 1);
                __builtin_prefetch(&Vg[(size_t)((kb + 2) * 32 + rkv) * 64 + ckv], 0, 1);
            }
        }

        // scores S = Q * K^T : two 16x16 tiles (keys t*16 .. t*16+15)
        v8f sc[2];
#pragma unroll
        for (int t = 0; t < 2; ++t) {
            FragAB b0, b1;   // B[k][n] = K[key=n][d = 32f + k]
            const int row = t * 16 + l15;
            const int sw  = swzQ(row);
            const unsigned short* rp = &Ks[cur][row * 64];
            b0.q[0] = lds128(rp + ((8 * hi) ^ sw));
            b0.q[1] = lds128(rp + ((16 + 8 * hi) ^ sw));
            b1.q[0] = lds128(rp + ((32 + 8 * hi) ^ sw));
            b1.q[1] = lds128(rp + ((32 + 16 + 8 * hi) ^ sw));
            v8f z;
#pragma unroll
            for (int r = 0; r < 8; ++r) z[r] = 0.0f;
            z = __builtin_amdgcn_wmma_f32_16x16x32_bf16(false, qa[0].v, false, b0.v,
                                                        (short)0, z, false, false);
            z = __builtin_amdgcn_wmma_f32_16x16x32_bf16(false, qa[1].v, false, b1.v,
                                                        (short)0, z, false, false);
            sc[t] = z;
        }

        // mask + scale (lane column = key index)
#pragma unroll
        for (int t = 0; t < 2; ++t) {
            const bool mv = mcur[t] != 0;
#pragma unroll
            for (int r = 0; r < 8; ++r)
                sc[t][r] = mv ? sc[t][r] * scale : NEG;
        }

        // online softmax: DPP row reductions (pure VALU, no LDS traffic)
#pragma unroll
        for (int r = 0; r < 8; ++r) {
            const float v  = rowmax16(fmaxf(sc[0][r], sc[1][r]));
            const float nm = fmaxf(mrow[r], v);
            const float alpha = __expf(mrow[r] - nm);
            mrow[r] = nm;
            const float p0 = __expf(sc[0][r] - nm);
            const float p1 = __expf(sc[1][r] - nm);
            sc[0][r] = p0; sc[1][r] = p1;
            lrow[r] = lrow[r] * alpha + rowsum16(p0 + p1);
            o[0][r] *= alpha; o[1][r] *= alpha; o[2][r] *= alpha; o[3][r] *= alpha;
        }

        // rearrange P from C/D accumulator layout to A-fragment layout via LDS
#pragma unroll
        for (int t = 0; t < 2; ++t)
#pragma unroll
            for (int r = 0; r < 8; ++r) {
                const int m = r + 8 * hi, col = t * 16 + l15;
                Ps[wave][m * 32 + (col ^ swzP(m))] = f2bf(sc[t][r]);
            }

        asm volatile("s_wait_dscnt 0" ::: "memory");   // wave-local LDS RAW fence

        FragAB pa;
        {
            const int sw = swzP(l15);
            const unsigned short* rp = &Ps[wave][l15 * 32];
            pa.q[0] = lds128(rp + ((8 * hi) ^ sw));
            pa.q[1] = lds128(rp + ((16 + 8 * hi) ^ sw));
        }

        // O += P * V : B[k][n] = V[key=k][d=n] read from VsT[d][key]
#pragma unroll
        for (int u = 0; u < 4; ++u) {
            FragAB bvf;
            const int d  = u * 16 + l15;
            const int sw = swzV(d);
            const unsigned short* rp = &VsT[cur][d * 32];
            bvf.q[0] = lds128(rp + ((8 * hi) ^ sw));
            bvf.q[1] = lds128(rp + ((16 + 8 * hi) ^ sw));
            o[u] = __builtin_amdgcn_wmma_f32_16x16x32_bf16(false, pa.v, false, bvf.v,
                                                           (short)0, o[u], false, false);
        }

        if (kb + 1 < NKB) {
            stageKV((kb + 1) & 1);
            mcur[0] = mnxt[0]; mcur[1] = mnxt[1];
        }
        __syncthreads();
    }

    // normalize + store ctx as (B*S, DIMK) bf16 row-major for the output GEMM
#pragma unroll
    for (int u = 0; u < 4; ++u)
#pragma unroll
        for (int r = 0; r < 8; ++r) {
            const float val = o[u][r] / lrow[r];
            const int s   = q0 + wave * 16 + r + 8 * hi;
            const int col = h * HDIM + u * 16 + l15;
            ctx[((size_t)b * SEQ + s) * DIMK + col] = f2bf(val);
        }
}

extern "C" void kernel_launch(void* const* d_in, const int* in_sizes, int n_in,
                              void* d_out, int out_size, void* d_ws, size_t ws_size,
                              hipStream_t stream) {
    (void)in_sizes; (void)n_in; (void)out_size; (void)ws_size;
    const float*         x   = (const float*)d_in[0];
    const unsigned char* msk = (const unsigned char*)d_in[1];
    const float* Wq = (const float*)d_in[2];
    const float* bq = (const float*)d_in[3];
    const float* Wk = (const float*)d_in[4];
    const float* bk = (const float*)d_in[5];
    const float* Wv = (const float*)d_in[6];
    const float* bv = (const float*)d_in[7];
    const float* Wo = (const float*)d_in[8];
    const float* bo = (const float*)d_in[9];
    float* out = (float*)d_out;

    char* ws = (char*)d_ws;
    const size_t MB = 1024 * 1024;
    unsigned short* xb  = (unsigned short*)(ws + 0);        // 8 MB (reused as ctx)
    unsigned short* Wqb = (unsigned short*)(ws +  8 * MB);  // 2 MB
    unsigned short* Wkb = (unsigned short*)(ws + 10 * MB);
    unsigned short* Wvb = (unsigned short*)(ws + 12 * MB);
    unsigned short* Wob = (unsigned short*)(ws + 14 * MB);
    unsigned short* Qb  = (unsigned short*)(ws + 16 * MB);  // 8 MB each
    unsigned short* Kb  = (unsigned short*)(ws + 24 * MB);
    unsigned short* Vb  = (unsigned short*)(ws + 32 * MB);
    unsigned short* ctx = xb;   // xb dead after QKV projections

    cvt_f32_bf16<<<2048, 256, 0, stream>>>(x,  xb,  MROWS * DIMK);
    cvt_f32_bf16<<<512,  256, 0, stream>>>(Wq, Wqb, DIMK * DIMK);
    cvt_f32_bf16<<<512,  256, 0, stream>>>(Wk, Wkb, DIMK * DIMK);
    cvt_f32_bf16<<<512,  256, 0, stream>>>(Wv, Wvb, DIMK * DIMK);
    cvt_f32_bf16<<<512,  256, 0, stream>>>(Wo, Wob, DIMK * DIMK);

    dim3 gg(MROWS / BM, DIMK / BN);   // 32 x 8
    gemm_bf16_wmma<0><<<gg, 256, 0, stream>>>(xb, Wqb, bq, Qb, nullptr);
    gemm_bf16_wmma<0><<<gg, 256, 0, stream>>>(xb, Wkb, bk, Kb, nullptr);
    gemm_bf16_wmma<0><<<gg, 256, 0, stream>>>(xb, Wvb, bv, Vb, nullptr);

    dim3 ga(SEQ / 128, BATCH * NHEADS);   // 16 x 32
    flash_attn_wmma<<<ga, 256, 0, stream>>>(Qb, Kb, Vb, msk, ctx);

    gemm_bf16_wmma<1><<<gg, 256, 0, stream>>>(ctx, Wob, bo, nullptr, out);
}
